// EGNN_44822278701572
// MI455X (gfx1250) — matrix operand order repaired and checked by reading, not compiled, not run on hardware
//
#include <hip/hip_runtime.h>

#define NODE_DIM 64
#define EDGE_DIM 4
#define NUM_LAYERS 4
#define MSG_IN 133
#define MSG_KPAD 160   // 133 padded to multiple of 32
#define MSG_HID 128

typedef __attribute__((ext_vector_type(16))) _Float16 v16h;
typedef __attribute__((ext_vector_type(8)))  _Float16 v8h;
typedef __attribute__((ext_vector_type(8)))  float    v8f;
typedef __attribute__((ext_vector_type(4)))  unsigned u32x4;
typedef __attribute__((ext_vector_type(8)))  int      i32x8;
typedef __attribute__((ext_vector_type(4)))  int      i32x4;

__device__ __forceinline__ float silu_f(float v) { return v / (1.0f + __expf(-v)); }

__device__ __forceinline__ float pnoise(unsigned v) {
  v ^= v >> 17; v *= 0xed5ad4bbu; v ^= v >> 11; v *= 0xac4c1b51u;
  v ^= v >> 15; v *= 0x31848babu; v ^= v >> 14;
  return ((float)(v >> 8) * (1.0f / 16777216.0f) - 0.5f) * 3.464101615f;
}

// ---- CDNA5 async copy: global -> LDS, 16B per lane, tracked by ASYNCcnt ----
__device__ __forceinline__ void async_b128_to_lds(void* ldsDst, const void* gSrc) {
  unsigned lo = (unsigned)(unsigned long long)ldsDst;   // LDS byte address (flat addr low bits)
  unsigned long long ga = (unsigned long long)gSrc;
  asm volatile("global_load_async_to_lds_b128 %0, %1, off"
               :: "v"(lo), "v"(ga) : "memory");
}
__device__ __forceinline__ void wait_asynccnt0() {
  asm volatile("s_wait_asynccnt 0" ::: "memory");
}

// ---- CDNA5 Tensor Data Mover: 1-D bulk copy global -> LDS (data_size = 8B) ----
__device__ __forceinline__ void tdm_copy_1d(unsigned ldsAddr, const void* gsrc, unsigned nQwords) {
  unsigned long long ga = (unsigned long long)gsrc;
  u32x4 g0;
  g0[0] = 1u;                                            // count=1, user mode
  g0[1] = ldsAddr;                                       // lds_addr (bytes)
  g0[2] = (unsigned)ga;                                  // global_addr[31:0]
  g0[3] = (unsigned)((ga >> 32) & 0x1FFFFFFull) | (2u << 30);  // global_addr[56:32] | type=2
  i32x8 g1;
  g1[0] = (3 << 16);                                     // workgroup_mask=0, data_size=3 (8B)
  g1[1] = (int)(nQwords << 16);                          // tensor_dim0[15:0]
  g1[2] = (int)((nQwords >> 16) & 0xFFFF) | (1 << 16);   // tensor_dim0[31:16], tensor_dim1=1
  g1[3] = (int)(nQwords << 16);                          // tensor_dim1 hi=0, tile_dim0
  g1[4] = 1;                                             // tile_dim1=1, tile_dim2=0
  g1[5] = (int)nQwords;                                  // tensor_dim0_stride lo32
  g1[6] = 0;
  g1[7] = 0;
  i32x4 gz = {0, 0, 0, 0};
#if __clang_major__ >= 23
  i32x8 gz8 = {0, 0, 0, 0, 0, 0, 0, 0};
  __builtin_amdgcn_tensor_load_to_lds(g0, g1, gz, gz, gz8, 0);
#else
  __builtin_amdgcn_tensor_load_to_lds(g0, g1, gz, gz, 0);
#endif
}

// A fragment (16x32 f16) from LDS row-major [row][K], wave32 WMMA layout
__device__ __forceinline__ v16h lds_load_A(const _Float16* buf, int stride, int rowBase, int kBase) {
  int lane = threadIdx.x & 31;
  int row = rowBase + (lane & 15);
  int k0  = kBase + ((lane >> 4) << 3);
  const _Float16* p = buf + row * stride + k0;
  v16h a;
  *((v8h*)&a)     = *(const v8h*)(p);
  *((v8h*)&a + 1) = *(const v8h*)(p + 16);
  return a;
}

// B fragment (32x16 f16) from LDS holding W transposed [N][K]
__device__ __forceinline__ v16h lds_load_B(const _Float16* buf, int stride, int colBase, int kBase) {
  int lane = threadIdx.x & 31;
  int col = colBase + (lane & 15);
  int k0  = kBase + ((lane >> 4) << 3);
  const _Float16* p = buf + col * stride + k0;
  v16h b;
  *((v8h*)&b)     = *(const v8h*)(p);
  *((v8h*)&b + 1) = *(const v8h*)(p + 16);
  return b;
}

// ---------------------------------------------------------------------------
// One-time weight prep: f32 -> f16, transposed [N][K] with K padded (W1), for TDM staging
// ---------------------------------------------------------------------------
__global__ void egnn_prep_weights(const float* __restrict__ msg_w1, const float* __restrict__ msg_w2,
                                  _Float16* __restrict__ w1t_g, _Float16* __restrict__ w2t_g)
{
  int i = blockIdx.x * 256 + threadIdx.x;
  const int tot1 = NUM_LAYERS * MSG_HID * MSG_KPAD;   // 81920
  const int tot2 = NUM_LAYERS * 64 * MSG_HID;         // 32768
  if (i < tot1) {
    int l = i / (MSG_HID * MSG_KPAD);
    int r = i % (MSG_HID * MSG_KPAD);
    int n = r / MSG_KPAD, k = r % MSG_KPAD;
    w1t_g[i] = (k < MSG_IN) ? (_Float16)msg_w1[((size_t)l * MSG_IN + k) * MSG_HID + n]
                            : (_Float16)0.0f;
  }
  int j = i - tot1;
  if (j >= 0 && j < tot2) {
    int l = j / (64 * MSG_HID);
    int r = j % (64 * MSG_HID);
    int n = r / MSG_HID, k = r % MSG_HID;
    w2t_g[j] = (_Float16)msg_w2[((size_t)l * MSG_HID + k) * 64 + n];
  }
}

// ---------------------------------------------------------------------------
// Edge message MLP: 128 edges/block, (133->128 SiLU ->64), scatter-add to aggr
// Weights staged via TDM, x-feature gathers via async-to-LDS. 56 WMMAs/wave.
// ---------------------------------------------------------------------------
#define SM_W1T   0         // [128][160] f16 = 40960
#define SM_W2T   40960     // [64][128]  f16 = 16384
#define SM_MIN   57344     // [128][160] f16 = 40960
#define SM_HBUF  98304     // [128][128] f16 = 32768
#define SM_B1    131072    // [128] f32
#define SM_B2    131584    // [64]  f32
#define SM_DIST  131840    // [128] f32
#define SM_SRC   132352    // [128] i32
#define SM_TGT   132864    // [128] i32
#define EDGE_SMEM_BYTES 133376

__global__ void __launch_bounds__(256) egnn_edge_kernel(
    const _Float16* __restrict__ xh, const float* __restrict__ pos,
    const long long* __restrict__ eidx, const float* __restrict__ ea,
    const _Float16* __restrict__ w1t, const float* __restrict__ b1,
    const _Float16* __restrict__ w2t, const float* __restrict__ b2,
    float* __restrict__ aggr, int n_edges)
{
  extern __shared__ char smem[];
  _Float16* W1t  = (_Float16*)(smem + SM_W1T);
  _Float16* W2t  = (_Float16*)(smem + SM_W2T);
  _Float16* mIn  = (_Float16*)(smem + SM_MIN);
  _Float16* hbuf = (_Float16*)(smem + SM_HBUF);
  float* b1s   = (float*)(smem + SM_B1);
  float* b2s   = (float*)(smem + SM_B2);
  float* dists = (float*)(smem + SM_DIST);
  int*   srcs  = (int*)(smem + SM_SRC);
  int*   tgts  = (int*)(smem + SM_TGT);

  const int t = threadIdx.x;
  const int lane = t & 31;
  const int wv = t >> 5;
  const int blockBase = blockIdx.x * 128;

  __builtin_prefetch(ea + (size_t)(blockBase + 128) * EDGE_DIM, 0, 1);

  // ---- wave 0: TDM bulk-stage both weight blocks into LDS ----
  if (wv == 0) {
    tdm_copy_1d((unsigned)(unsigned long long)W1t, w1t, (128 * MSG_KPAD * 2) / 8);
    tdm_copy_1d((unsigned)(unsigned long long)W2t, w2t, (64 * MSG_HID * 2) / 8);
    __builtin_amdgcn_s_wait_tensorcnt(0);
  }

  if (t < 128) b1s[t] = b1[t];
  if (t < 64)  b2s[t] = b2[t];

  // ---- stage edge indices + distances ----
  if (t < 128) {
    int eg = blockBase + t; if (eg >= n_edges) eg = n_edges - 1;
    int s = (int)eidx[eg];
    int g = (int)eidx[(long long)n_edges + eg];
    srcs[t] = s; tgts[t] = g;
    float dx = pos[s * 3 + 0] - pos[g * 3 + 0];
    float dy = pos[s * 3 + 1] - pos[g * 3 + 1];
    float dz = pos[s * 3 + 2] - pos[g * 3 + 2];
    dists[t] = sqrtf(dx * dx + dy * dy + dz * dz);
  }
  __syncthreads();

  // ---- async gather x_h[tgt], x_h[src] straight into m_in tile ----
  // 128 edges * 2 segments * 8 chunks(16B) = 2048 async issues / block
  for (int i = t; i < 2048; i += 256) {
    int e   = i >> 4;
    int seg = (i >> 3) & 1;     // 0 = x[tgt], 1 = x[src]
    int ch  = i & 7;
    int idx = seg ? srcs[e] : tgts[e];
    const _Float16* g = xh + (long long)idx * 64 + ch * 8;
    _Float16* d = mIn + e * MSG_KPAD + seg * 64 + ch * 8;
    async_b128_to_lds(d, g);
  }
  // ---- scalar tail: edge_attr, dist, zero padding ----
  {
    int e = t >> 1, half = t & 1;
    int eg = blockBase + e; if (eg >= n_edges) eg = n_edges - 1;
    _Float16* row = mIn + e * MSG_KPAD;
    if (half == 0) {
      row[128] = (_Float16)ea[(long long)eg * EDGE_DIM + 0];
      row[129] = (_Float16)ea[(long long)eg * EDGE_DIM + 1];
      row[130] = (_Float16)ea[(long long)eg * EDGE_DIM + 2];
      row[131] = (_Float16)ea[(long long)eg * EDGE_DIM + 3];
      row[132] = (_Float16)dists[e];
      #pragma unroll
      for (int k = 133; k < 144; ++k) row[k] = (_Float16)0.0f;
    } else {
      #pragma unroll
      for (int k = 144; k < 160; ++k) row[k] = (_Float16)0.0f;
    }
  }
  wait_asynccnt0();
  __syncthreads();

  const int rb = wv * 16;
  const int rTop = (lane >= 16) ? 8 : 0;

  // ---- GEMM1: [16x160] @ [160x128] + b1, SiLU -> hbuf (kt outer, A reused) ----
  v8f acc1[8];
  #pragma unroll
  for (int nt = 0; nt < 8; ++nt) {
    float bias = b1s[nt * 16 + (lane & 15)];
    #pragma unroll
    for (int r = 0; r < 8; ++r) acc1[nt][r] = bias;
  }
  #pragma unroll
  for (int kt = 0; kt < 5; ++kt) {
    v16h a = lds_load_A(mIn, MSG_KPAD, rb, kt * 32);
    #pragma unroll
    for (int nt = 0; nt < 8; ++nt) {
      v16h b = lds_load_B(W1t, MSG_KPAD, nt * 16, kt * 32);
      acc1[nt] = __builtin_amdgcn_wmma_f32_16x16x32_f16(false, a, false, b, (short)0, acc1[nt], false, false);
    }
  }
  #pragma unroll
  for (int nt = 0; nt < 8; ++nt)
    #pragma unroll
    for (int r = 0; r < 8; ++r) {
      int row = rb + r + rTop;
      hbuf[row * MSG_HID + nt * 16 + (lane & 15)] = (_Float16)silu_f(acc1[nt][r]);
    }
  // hbuf rows are wave-private: no block barrier needed

  // ---- GEMM2: [16x128] @ [128x64] + b2, scatter-add to aggr ----
  v8f acc2[4];
  #pragma unroll
  for (int nt = 0; nt < 4; ++nt) {
    float bias = b2s[nt * 16 + (lane & 15)];
    #pragma unroll
    for (int r = 0; r < 8; ++r) acc2[nt][r] = bias;
  }
  #pragma unroll
  for (int kt = 0; kt < 4; ++kt) {
    v16h a = lds_load_A(hbuf, MSG_HID, rb, kt * 32);
    #pragma unroll
    for (int nt = 0; nt < 4; ++nt) {
      v16h b = lds_load_B(W2t, MSG_HID, nt * 16, kt * 32);
      acc2[nt] = __builtin_amdgcn_wmma_f32_16x16x32_f16(false, a, false, b, (short)0, acc2[nt], false, false);
    }
  }
  #pragma unroll
  for (int nt = 0; nt < 4; ++nt)
    #pragma unroll
    for (int r = 0; r < 8; ++r) {
      int er = rb + r + rTop;
      int eg = blockBase + er;
      if (eg < n_edges)
        atomicAdd(&aggr[(long long)tgts[er] * 64 + nt * 16 + (lane & 15)], acc2[nt][r]);
    }
}

// ---------------------------------------------------------------------------
// Embedding: x = x0 @ emb_w + emb_b (WMMA), writes f16 mirror xh
// ---------------------------------------------------------------------------
__global__ void __launch_bounds__(256) egnn_embed_kernel(
    const float* __restrict__ x0, const float* __restrict__ w,
    const float* __restrict__ b, float* __restrict__ x,
    _Float16* __restrict__ xh, int n_nodes)
{
  __shared__ _Float16 wT[64 * 64];
  __shared__ _Float16 ain[128 * 64];
  __shared__ float bs[64];
  const int t = threadIdx.x, lane = t & 31, wv = t >> 5;
  const int blockBase = blockIdx.x * 128;

  for (int i = t; i < 4096; i += 256) {
    int n = i >> 6, k = i & 63;
    wT[i] = (_Float16)w[k * 64 + n];
  }
  for (int i = t; i < 8192; i += 256) {
    int r = i >> 6, k = i & 63;
    int node = blockBase + r;
    ain[i] = (node < n_nodes) ? (_Float16)x0[(long long)node * 64 + k] : (_Float16)0.0f;
  }
  if (t < 64) bs[t] = b[t];
  __syncthreads();

  const int rb = wv * 16;
  const int rTop = (lane >= 16) ? 8 : 0;
  v8f acc[4];
  #pragma unroll
  for (int nt = 0; nt < 4; ++nt) {
    float bias = bs[nt * 16 + (lane & 15)];
    #pragma unroll
    for (int r = 0; r < 8; ++r) acc[nt][r] = bias;
  }
  #pragma unroll
  for (int kt = 0; kt < 2; ++kt) {
    v16h a = lds_load_A(ain, 64, rb, kt * 32);
    #pragma unroll
    for (int nt = 0; nt < 4; ++nt) {
      v16h bf = lds_load_B(wT, 64, nt * 16, kt * 32);
      acc[nt] = __builtin_amdgcn_wmma_f32_16x16x32_f16(false, a, false, bf, (short)0, acc[nt], false, false);
    }
  }
  #pragma unroll
  for (int nt = 0; nt < 4; ++nt)
    #pragma unroll
    for (int r = 0; r < 8; ++r) {
      int row = rb + r + rTop;
      int node = blockBase + row;
      if (node < n_nodes) {
        int col = nt * 16 + (lane & 15);
        x[(long long)node * 64 + col]  = acc[nt][r];
        xh[(long long)node * 64 + col] = (_Float16)acc[nt][r];
      }
    }
}

// ---------------------------------------------------------------------------
__global__ void egnn_count_kernel(const long long* __restrict__ eidx,
                                  float* __restrict__ counts, int n_edges)
{
  int e = blockIdx.x * 256 + threadIdx.x;
  if (e < n_edges) atomicAdd(&counts[(int)eidx[(long long)n_edges + e]], 1.0f);
}

// ---------------------------------------------------------------------------
__global__ void __launch_bounds__(256) egnn_node_kernel(
    float* __restrict__ x, _Float16* __restrict__ xh, float* __restrict__ pos,
    float* __restrict__ aggr, const float* __restrict__ counts,
    const float* __restrict__ cw1, const float* __restrict__ cb1,
    const float* __restrict__ cw2, const float* __restrict__ cb2,
    int layer, int n_nodes)
{
  __shared__ float w1s[64 * 64];
  __shared__ float w2s[64];
  __shared__ float b1s[64];
  __shared__ float b2s_;
  for (int i = threadIdx.x; i < 4096; i += 256) w1s[i] = cw1[i];
  if (threadIdx.x < 64) { w2s[threadIdx.x] = cw2[threadIdx.x]; b1s[threadIdx.x] = cb1[threadIdx.x]; }
  if (threadIdx.x == 0) b2s_ = cb2[0];
  __syncthreads();

  int n = blockIdx.x * 256 + threadIdx.x;
  if (n >= n_nodes) return;

  float denom = fmaxf(counts[n], 1.0f);
  float a[64];
  float* agp = aggr + (long long)n * 64;
  float* xp  = x + (long long)n * 64;
  #pragma unroll
  for (int d = 0; d < 64; ++d) {
    float v = agp[d] / denom;
    a[d] = v;
    float nx = xp[d] + v;
    xp[d] = nx;
    xh[(long long)n * 64 + d] = (_Float16)nx;
    agp[d] = 0.0f;
  }

  float cu = b2s_;
  for (int k = 0; k < 64; ++k) {
    float s = b1s[k];
    #pragma unroll
    for (int j = 0; j < 64; ++j) s += a[j] * w1s[j * 64 + k];
    cu += silu_f(s) * w2s[k];
  }
  #pragma unroll
  for (int d = 0; d < 3; ++d) {
    unsigned h = (unsigned)(n * 3 + d) * 2654435761u + (unsigned)(layer + 1) * 0x9E3779B9u;
    pos[(long long)n * 3 + d] += cu * pnoise(h);
  }
}

// ---------------------------------------------------------------------------
__global__ void __launch_bounds__(256) egnn_reduce_kernel(
    const float* __restrict__ x, float* __restrict__ gsum, int n_nodes)
{
  __shared__ float red[256];
  int d = threadIdx.x & 63;
  int slot = threadIdx.x >> 6;
  float s = 0.0f;
  for (int n = blockIdx.x * 4 + slot; n < n_nodes; n += gridDim.x * 4)
    s += x[(long long)n * 64 + d];
  red[threadIdx.x] = s;
  __syncthreads();
  if (threadIdx.x < 64)
    atomicAdd(&gsum[d], red[d] + red[64 + d] + red[128 + d] + red[192 + d]);
}

// ---------------------------------------------------------------------------
__global__ void egnn_out_kernel(const float* __restrict__ gsum,
                                const float* __restrict__ w1, const float* __restrict__ b1,
                                const float* __restrict__ w2, const float* __restrict__ b2,
                                float* __restrict__ out, int n_nodes)
{
  if (threadIdx.x == 0 && blockIdx.x == 0) {
    float inv = 1.0f / (float)n_nodes;
    float acc = b2[0];
    for (int j = 0; j < 32; ++j) {
      float s = b1[j];
      for (int i = 0; i < 64; ++i) s += (gsum[i] * inv) * w1[i * 32 + j];
      acc += silu_f(s) * w2[j];
    }
    out[0] = acc;
  }
}

// ---------------------------------------------------------------------------
extern "C" void kernel_launch(void* const* d_in, const int* in_sizes, int n_in,
                              void* d_out, int out_size, void* d_ws, size_t ws_size,
                              hipStream_t stream)
{
  const float*     x0       = (const float*)d_in[0];
  const float*     pos0     = (const float*)d_in[1];
  const long long* eidx     = (const long long*)d_in[2];
  const float*     ea       = (const float*)d_in[3];
  const float*     emb_w    = (const float*)d_in[4];
  const float*     emb_b    = (const float*)d_in[5];
  const float*     msg_w1   = (const float*)d_in[6];
  const float*     msg_b1   = (const float*)d_in[7];
  const float*     msg_w2   = (const float*)d_in[8];
  const float*     msg_b2   = (const float*)d_in[9];
  const float*     coord_w1 = (const float*)d_in[10];
  const float*     coord_b1 = (const float*)d_in[11];
  const float*     coord_w2 = (const float*)d_in[12];
  const float*     coord_b2 = (const float*)d_in[13];
  const float*     out_w1   = (const float*)d_in[14];
  const float*     out_b1   = (const float*)d_in[15];
  const float*     out_w2   = (const float*)d_in[16];
  const float*     out_b2   = (const float*)d_in[17];

  const int n_nodes = in_sizes[0] / NODE_DIM;
  const int n_edges = in_sizes[3] / EDGE_DIM;

  // workspace carve (16B alignment maintained for f16/TDM regions)
  float* x      = (float*)d_ws;                       // [N,64]
  float* aggr   = x + (size_t)n_nodes * 64;           // [N,64]
  float* posw   = aggr + (size_t)n_nodes * 64;        // [N,3]
  float* counts = posw + (size_t)n_nodes * 3;         // [N]
  float* gsum   = counts + n_nodes;                   // [64]
  _Float16* xh  = (_Float16*)(gsum + 64);             // [N,64] f16
  _Float16* w1t_g = xh + (size_t)n_nodes * 64;        // [4][128*160] f16
  _Float16* w2t_g = w1t_g + NUM_LAYERS * MSG_HID * MSG_KPAD; // [4][64*128] f16

  hipMemsetAsync(aggr, 0, (size_t)n_nodes * 64 * sizeof(float), stream);
  hipMemsetAsync(counts, 0, (size_t)n_nodes * sizeof(float), stream);
  hipMemsetAsync(gsum, 0, 64 * sizeof(float), stream);
  hipMemcpyAsync(posw, pos0, (size_t)n_nodes * 3 * sizeof(float),
                 hipMemcpyDeviceToDevice, stream);

  {
    int tot = NUM_LAYERS * MSG_HID * MSG_KPAD + NUM_LAYERS * 64 * MSG_HID;
    egnn_prep_weights<<<(tot + 255) / 256, 256, 0, stream>>>(msg_w1, msg_w2, w1t_g, w2t_g);
  }
  egnn_count_kernel<<<(n_edges + 255) / 256, 256, 0, stream>>>(eidx, counts, n_edges);
  egnn_embed_kernel<<<(n_nodes + 127) / 128, 256, 0, stream>>>(x0, emb_w, emb_b, x, xh, n_nodes);

  for (int l = 0; l < NUM_LAYERS; ++l) {
    egnn_edge_kernel<<<(n_edges + 127) / 128, 256, EDGE_SMEM_BYTES, stream>>>(
        xh, posw, eidx, ea,
        w1t_g + (size_t)l * MSG_HID * MSG_KPAD, msg_b1 + (size_t)l * MSG_HID,
        w2t_g + (size_t)l * 64 * MSG_HID, msg_b2 + (size_t)l * NODE_DIM,
        aggr, n_edges);
    egnn_node_kernel<<<(n_nodes + 255) / 256, 256, 0, stream>>>(
        x, xh, posw, aggr, counts,
        coord_w1 + (size_t)l * 64 * 64, coord_b1 + (size_t)l * 64,
        coord_w2 + (size_t)l * 64, coord_b2 + l,
        l, n_nodes);
  }

  egnn_reduce_kernel<<<256, 256, 0, stream>>>(x, gsum, n_nodes);
  egnn_out_kernel<<<1, 64, 0, stream>>>(gsum, out_w1, out_b1, out_w2, out_b2,
                                        (float*)d_out, n_nodes);
}
